// Model_13786845020729
// MI455X (gfx1250) — compile-verified, matrix operand order checked by reading, not run on hardware
//
#include <hip/hip_runtime.h>
#include <math.h>

// ---------------------------------------------------------------------------
// Problem constants (from the reference)
// ---------------------------------------------------------------------------
#define Bb    16
#define Ll    512
#define Vv    32
#define Dd    128
#define DFFc  256
#define PLc   16
#define STc   8
#define PREDc 96
#define DIc   256
#define DSc   16
#define DTRc  8
#define Hh    8
#define HDc   32
#define Kc    4
#define Pp    64          // (L-PL)/ST + 2
#define NFc   8192        // D*P
#define RT    32768       // B*V*P token rows
#define RC    512         // B*V channel rows
#define ZXW   552         // 2*DI + 2*DS + H
#define XBCW  288         // DI + 2*DS
#define WXW   40          // DTR + 2*DS

typedef __attribute__((ext_vector_type(8)))  _Float16 v8h;
typedef __attribute__((ext_vector_type(16))) _Float16 v16h;
typedef __attribute__((ext_vector_type(8)))  float    v8f;

#define DEV __device__ __forceinline__

// Optional CDNA5 async global->LDS path (ASYNCcnt-tracked), feature-gated so
// the file compiles on toolchains without the builtins.
#if defined(__has_builtin)
#if __has_builtin(__builtin_amdgcn_global_load_async_to_lds_b32) && \
    __has_builtin(__builtin_amdgcn_s_wait_asynccnt)
#define HAVE_ASYNC_LDS 1
#endif
#endif

typedef __attribute__((address_space(1))) int glb_i32;
typedef __attribute__((address_space(3))) int lds_i32;

DEV float siluf(float x) { return x / (1.f + __expf(-x)); }
DEV float geluf(float x) {
  float x3 = x * x * x;
  return 0.5f * x * (1.f + tanhf(0.7978845608028654f * (x + 0.044715f * x3)));
}
DEV float softplusf(float x) { return (x > 20.f) ? x : log1pf(__expf(x)); }

// ---------------------------------------------------------------------------
// Instance norm over L per (b,v): means/stdev + normalized xc (f32 + f16)
// ---------------------------------------------------------------------------
__global__ void k_norm(const float* __restrict__ x_enc, float* __restrict__ xc32,
                       _Float16* __restrict__ xc16, float* __restrict__ mean,
                       float* __restrict__ stdev) {
  int bv = blockIdx.x, b = bv >> 5, v = bv & 31;
  int t = threadIdx.x;
  const float* xp = x_enc + (size_t)b * Ll * Vv + v;
  float a0 = xp[(size_t)t * Vv];
  float a1 = xp[(size_t)(t + 256) * Vv];
  __shared__ float ss[256], sq[256];
  ss[t] = a0 + a1;
  sq[t] = a0 * a0 + a1 * a1;
  __syncthreads();
  for (int off = 128; off > 0; off >>= 1) {
    if (t < off) { ss[t] += ss[t + off]; sq[t] += sq[t + off]; }
    __syncthreads();
  }
  __shared__ float m_, s_;
  if (t == 0) {
    float m = ss[0] * (1.f / Ll);
    float var = sq[0] * (1.f / Ll) - m * m;
    float sd = sqrtf(var + 1e-5f);
    m_ = m; s_ = sd;
    mean[bv] = m; stdev[bv] = sd;
  }
  __syncthreads();
  float m = m_, inv = 1.f / s_;
  float y0 = (a0 - m) * inv, y1 = (a1 - m) * inv;
  size_t base = (size_t)bv * Ll;
  xc32[base + t] = y0;           xc32[base + t + 256] = y1;
  xc16[base + t] = (_Float16)y0; xc16[base + t + 256] = (_Float16)y1;
}

// ---------------------------------------------------------------------------
// f32 -> f16 converter (weights / activations)
// ---------------------------------------------------------------------------
__global__ void k_cvt(const float* __restrict__ in, _Float16* __restrict__ out, int n) {
  int i = blockIdx.x * 256 + threadIdx.x;
  if (i < n) out[i] = (_Float16)in[i];
}

// ---------------------------------------------------------------------------
// Patch embedding (K=16, too small for WMMA): staged via LDS.
// ---------------------------------------------------------------------------
__global__ void k_patch(const float* __restrict__ xc32, const float* __restrict__ Wp,
                        const float* __restrict__ bp, _Float16* __restrict__ tw16) {
  int row = blockIdx.x;              // bv*64 + p
  int bv = row >> 6, p = row & 63;
  int d = threadIdx.x;               // 128
  __shared__ float sp[PLc];
  if (d < PLc) {
    int l = p * STc + d;
    if (l > Ll - 1) l = Ll - 1;      // edge pad
    sp[d] = xc32[(size_t)bv * Ll + l];
  }
  __syncthreads();
  float acc = bp[d];
#pragma unroll
  for (int j = 0; j < PLc; ++j) acc += sp[j] * Wp[d * PLc + j];
  tw16[(size_t)row * Dd + d] = (_Float16)acc;
}

// ---------------------------------------------------------------------------
// WMMA GEMM: C[M,N] = act( A[M,K] @ Bw[N,K]^T + bias ) (+ res)
// One wave owns a 16x64 tile: a single A fragment feeds 4 WMMAs per K-step
// (4x A-load amortization, ~205 FLOP/B). 4 waves/block stack along M, so M
// must be a multiple of 64 (true for all call sites: 32768, 512).
// Ragged N: clamped B loads + masked stores; all branches wave-uniform so
// EXEC is all-ones around v_wmma as the ISA requires.  K multiple of 32.
// act: 0 = none, 1 = gelu(tanh)
// ---------------------------------------------------------------------------
__global__ void k_wmma_gemm(const _Float16* __restrict__ A, const _Float16* __restrict__ Bw,
                            const float* __restrict__ bias, const float* __restrict__ res,
                            float* __restrict__ C, _Float16* __restrict__ C16,
                            int M, int N, int K, int act) {
  int lane = threadIdx.x & 31;
  int wave = threadIdx.x >> 5;
  int tile_m = blockIdx.x * 4 + wave;     // 16-row tile
  int ngrp   = blockIdx.y;                // 64-col group
  if (ngrp * 64 >= N) return;             // wave-uniform
  int half = lane >> 4, lm = lane & 15;
  int arow = tile_m * 16 + lm;            // in range: M multiple of 64
  const _Float16* Ap = A + (size_t)arow * K + (half << 3);
  const _Float16* Bp[4];
#pragma unroll
  for (int s = 0; s < 4; ++s) {
    int brow = ngrp * 64 + s * 16 + lm;
    if (brow >= N) brow = N - 1;          // clamp; garbage columns masked on store
    Bp[s] = Bw + (size_t)brow * K + (half << 3);
  }
  v8f zero = {};
  v8f acc[4] = {zero, zero, zero, zero};
  for (int k0 = 0; k0 < K; k0 += 32) {
    v8h a0 = *(const v8h*)(Ap + k0);
    v8h a1 = *(const v8h*)(Ap + k0 + 16);
    __builtin_prefetch(Ap + k0 + 64, 0, 0);      // global_prefetch_b8 (speculative)
    v16h a = __builtin_shufflevector(a0, a1, 0, 1, 2, 3, 4, 5, 6, 7,
                                     8, 9, 10, 11, 12, 13, 14, 15);
#pragma unroll
    for (int s = 0; s < 4; ++s) {
      v8h b0 = *(const v8h*)(Bp[s] + k0);
      v8h b1 = *(const v8h*)(Bp[s] + k0 + 16);
      __builtin_prefetch(Bp[s] + k0 + 64, 0, 0);
      v16h b = __builtin_shufflevector(b0, b1, 0, 1, 2, 3, 4, 5, 6, 7,
                                       8, 9, 10, 11, 12, 13, 14, 15);
      acc[s] = __builtin_amdgcn_wmma_f32_16x16x32_f16(false, a, false, b,
                                                      (short)0, acc[s], false, false);
    }
  }
#pragma unroll
  for (int s = 0; s < 4; ++s) {
    int n = ngrp * 64 + s * 16 + lm;
    if (n < N) {
      float bv = bias ? bias[n] : 0.f;
#pragma unroll
      for (int r = 0; r < 8; ++r) {
        int m = tile_m * 16 + half * 8 + r;   // C/D fragment: VGPR r -> row r + 8*half
        float x = acc[s][r] + bv;
        if (act == 1) x = geluf(x);
        size_t idx = (size_t)m * N + n;
        if (res) x += res[idx];
        C[idx] = x;
        if (C16) C16[idx] = (_Float16)x;
      }
    }
  }
}

// ---------------------------------------------------------------------------
// Mamba depthwise causal conv (K=4) along p + bias + silu.
// ---------------------------------------------------------------------------
__global__ void k_dconv_m(const float* __restrict__ mbin, const float* __restrict__ w,
                          const float* __restrict__ bcv, float* __restrict__ out32,
                          _Float16* __restrict__ out16) {
  int row = blockIdx.x, p = row & 63;
  int c = threadIdx.x;               // 256
  float acc = bcv[c];
#pragma unroll
  for (int j = 0; j < Kc; ++j)
    if (p - j >= 0)
      acc += mbin[(size_t)(row - j) * (2 * DIc) + c] * w[c * Kc + (Kc - 1 - j)];
  acc = siluf(acc);
  out32[(size_t)row * DIc + c] = acc;
  out16[(size_t)row * DIc + c] = (_Float16)acc;
}

// ---------------------------------------------------------------------------
// dt projection: softplus(dtraw(8) @ Wdt^T + bdt), K=8 -> VALU.
// ---------------------------------------------------------------------------
__global__ void k_dtproj(const float* __restrict__ wx, const float* __restrict__ Wdt,
                         const float* __restrict__ bdt, float* __restrict__ dt) {
  int row = blockIdx.x;
  int c = threadIdx.x;               // 256
  __shared__ float sd[DTRc];
  if (c < DTRc) sd[c] = wx[(size_t)row * WXW + c];
  __syncthreads();
  float acc = bdt[c];
#pragma unroll
  for (int j = 0; j < DTRc; ++j) acc += sd[j] * Wdt[c * DTRc + j];
  dt[(size_t)row * DIc + c] = softplusf(acc);
}

// ---------------------------------------------------------------------------
// Mamba selective scan: one thread per (bv, channel), 16 states in VGPRs,
// 64 sequential steps. Per-step B/C coefficients (32 contiguous floats at
// wx[row*40+8]) are broadcast into LDS — via the CDNA5 async global->LDS
// path when available (ASYNCcnt), else plain stores.
// ---------------------------------------------------------------------------
__global__ void k_scan_m(const float* __restrict__ xm, const float* __restrict__ dt,
                         const float* __restrict__ wx, const float* __restrict__ mbin,
                         const float* __restrict__ Alog, const float* __restrict__ Dp,
                         _Float16* __restrict__ yg) {
  int bv = blockIdx.x;
  int c = threadIdx.x;               // 256
  float Acf[DSc];
#pragma unroll
  for (int s = 0; s < DSc; ++s) Acf[s] = -__expf(Alog[c * DSc + s]);
  float Dc = Dp[c];
  float h[DSc];
#pragma unroll
  for (int s = 0; s < DSc; ++s) h[s] = 0.f;
  __shared__ float sBC[2 * DSc];     // [0,16): B   [16,32): C
  for (int t = 0; t < Pp; ++t) {
    int row = bv * Pp + t;
    __syncthreads();
#ifdef HAVE_ASYNC_LDS
    if (c < 2 * DSc) {               // wave 0 only: full-wave uniform issue
      const float* g = wx + (size_t)row * WXW + DTRc + c;
      __builtin_amdgcn_global_load_async_to_lds_b32(
          (glb_i32*)(unsigned long long)(uintptr_t)g,
          (lds_i32*)(unsigned)(uintptr_t)&sBC[c],
          0, 0);
    }
    __builtin_amdgcn_s_wait_asynccnt(0);
    __syncthreads();
#else
    if (c < 2 * DSc) sBC[c] = wx[(size_t)row * WXW + DTRc + c];
    __syncthreads();
#endif
    float dtv = dt[(size_t)row * DIc + c];
    float xv  = xm[(size_t)row * DIc + c];
    float dx  = dtv * xv;
    float y = 0.f;
#pragma unroll
    for (int s = 0; s < DSc; ++s) {
      h[s] = __expf(dtv * Acf[s]) * h[s] + dx * sBC[s];
      y += h[s] * sBC[DSc + s];
    }
    y += Dc * xv;
    float zv = mbin[(size_t)row * (2 * DIc) + DIc + c];
    yg[(size_t)row * DIc + c] = (_Float16)(y * siluf(zv));
  }
}

// ---------------------------------------------------------------------------
// Hybrid-path depthwise conv over v (seq len 32) on zx cols [256,544) + silu.
// ---------------------------------------------------------------------------
__global__ void k_dconv_h(const float* __restrict__ zx, const float* __restrict__ w,
                          const float* __restrict__ bcv, float* __restrict__ xbc) {
  int row = blockIdx.x, v = row & 31;
  int c = threadIdx.x;               // 288
  float acc = bcv[c];
#pragma unroll
  for (int j = 0; j < Kc; ++j)
    if (v - j >= 0)
      acc += zx[(size_t)(row - j) * ZXW + DIc + c] * w[c * Kc + (Kc - 1 - j)];
  xbc[(size_t)row * XBCW + c] = siluf(acc);
}

// ---------------------------------------------------------------------------
// SSD forward+backward scan over v. One thread per (b, h, hd).
// ---------------------------------------------------------------------------
__global__ void k_hy_scan(const float* __restrict__ zx, const float* __restrict__ xbc,
                          const float* __restrict__ Alog, const float* __restrict__ bdt,
                          const float* __restrict__ Dp, float* __restrict__ yh) {
  int b = blockIdx.x;
  int h = threadIdx.x >> 5, hd = threadIdx.x & 31;
  float Ah = -__expf(Alog[h]);
  float Dh = Dp[h];
  float bd = bdt[h];
  float st[DSc];
  float yf[Vv];
#pragma unroll
  for (int s = 0; s < DSc; ++s) st[s] = 0.f;
#pragma unroll
  for (int v = 0; v < Vv; ++v) {
    size_t row = (size_t)(b * Vv + v);
    float dtv = softplusf(zx[row * ZXW + 2 * DIc + 2 * DSc + h] + bd);
    float e = __expf(dtv * Ah);
    float xv = xbc[row * XBCW + h * HDc + hd];
    float dx = dtv * xv;
    float y = 0.f;
#pragma unroll
    for (int s = 0; s < DSc; ++s) {
      st[s] = e * st[s] + dx * xbc[row * XBCW + DIc + s];
      y += st[s] * xbc[row * XBCW + DIc + DSc + s];
    }
    yf[v] = y;
  }
#pragma unroll
  for (int s = 0; s < DSc; ++s) st[s] = 0.f;
#pragma unroll
  for (int v = Vv - 1; v >= 0; --v) {
    size_t row = (size_t)(b * Vv + v);
    float dtv = softplusf(zx[row * ZXW + 2 * DIc + 2 * DSc + h] + bd);
    float e = __expf(dtv * Ah);
    float xv = xbc[row * XBCW + h * HDc + hd];
    float dx = dtv * xv;
    float y = 0.f;
#pragma unroll
    for (int s = 0; s < DSc; ++s) {
      st[s] = e * st[s] + dx * xbc[row * XBCW + DIc + s];
      y += st[s] * xbc[row * XBCW + DIc + DSc + s];
    }
    float tot = yf[v] + y + Dh * xv;
    float zv = zx[row * ZXW + h * HDc + hd];   // zh gate
    yh[row * DIc + h * HDc + hd] = tot * siluf(zv);
  }
}

// ---------------------------------------------------------------------------
// RMS norm over 256 cols * hy_normw -> f16
// ---------------------------------------------------------------------------
__global__ void k_rms(const float* __restrict__ yh, const float* __restrict__ nw,
                      _Float16* __restrict__ out16) {
  int row = blockIdx.x;
  int c = threadIdx.x;               // 256
  float v = yh[(size_t)row * DIc + c];
  __shared__ float ss[256];
  ss[c] = v * v;
  __syncthreads();
  for (int off = 128; off > 0; off >>= 1) {
    if (c < off) ss[c] += ss[c + off];
    __syncthreads();
  }
  __shared__ float scale_;
  if (c == 0) scale_ = rsqrtf(ss[0] * (1.f / DIc) + 1e-5f);
  __syncthreads();
  out16[(size_t)row * DIc + c] = (_Float16)(v * scale_ * nw[c]);
}

// ---------------------------------------------------------------------------
// FiLM fuse: fused[row, d*64+p] = gamma[row,d]*tw_enc[row*64+p, d] + beta[row,d]
// ---------------------------------------------------------------------------
__global__ void k_fuse(const float* __restrict__ film, const float* __restrict__ twe,
                       _Float16* __restrict__ fused) {
  int idx = blockIdx.x * 256 + threadIdx.x;   // exact grid: 512*8192
  int row = idx >> 13, f = idx & 8191;
  int d = f >> 6, p = f & 63;
  float g  = film[(size_t)row * 256 + d];
  float be = film[(size_t)row * 256 + 128 + d];
  float tv = twe[((size_t)row * Pp + p) * Dd + d];
  fused[idx] = (_Float16)(g * tv + be);
}

// ---------------------------------------------------------------------------
// De-norm + transpose to (B, PRED, V)
// ---------------------------------------------------------------------------
__global__ void k_final(const float* __restrict__ ho, const float* __restrict__ mean,
                        const float* __restrict__ stdev, float* __restrict__ out) {
  int idx = blockIdx.x * 256 + threadIdx.x;
  if (idx >= RC * PREDc) return;
  int row = idx / PREDc, t = idx - row * PREDc;
  int b = row >> 5, v = row & 31;
  out[(size_t)b * PREDc * Vv + (size_t)t * Vv + v] = ho[idx] * stdev[row] + mean[row];
}

// ---------------------------------------------------------------------------
// Host orchestration
// ---------------------------------------------------------------------------
extern "C" void kernel_launch(void* const* d_in, const int* in_sizes, int n_in,
                              void* d_out, int out_size, void* d_ws, size_t ws_size,
                              hipStream_t stream) {
  (void)in_sizes; (void)n_in; (void)out_size; (void)ws_size;
  const float* x_enc    = (const float*)d_in[0];
  const float* W_patch  = (const float*)d_in[4];
  const float* b_patch  = (const float*)d_in[5];
  const float* W_chan   = (const float*)d_in[6];
  const float* b_chan   = (const float*)d_in[7];
  const float* mb_Win   = (const float*)d_in[8];
  const float* mb_conv  = (const float*)d_in[9];
  const float* mb_convb = (const float*)d_in[10];
  const float* mb_Wx    = (const float*)d_in[11];
  const float* mb_Wdt   = (const float*)d_in[12];
  const float* mb_bdt   = (const float*)d_in[13];
  const float* mb_Alog  = (const float*)d_in[14];
  const float* mb_D     = (const float*)d_in[15];
  const float* mb_Wout  = (const float*)d_in[16];
  const float* tf_W1    = (const float*)d_in[17];
  const float* tf_b1    = (const float*)d_in[18];
  const float* tf_W2    = (const float*)d_in[19];
  const float* tf_b2    = (const float*)d_in[20];
  const float* hy_Win   = (const float*)d_in[21];
  const float* hy_conv  = (const float*)d_in[22];
  const float* hy_convb = (const float*)d_in[23];
  const float* hy_bdt   = (const float*)d_in[24];
  const float* hy_Alog  = (const float*)d_in[25];
  const float* hy_D     = (const float*)d_in[26];
  const float* hy_normw = (const float*)d_in[27];
  const float* hy_Wout  = (const float*)d_in[28];
  const float* cf_W1    = (const float*)d_in[29];
  const float* cf_b1    = (const float*)d_in[30];
  const float* cf_W2    = (const float*)d_in[31];
  const float* cf_b2    = (const float*)d_in[32];
  const float* film_W   = (const float*)d_in[33];
  const float* film_b   = (const float*)d_in[34];
  const float* head_W   = (const float*)d_in[35];
  const float* head_b   = (const float*)d_in[36];
  float* out = (float*)d_out;

  char* ws = (char*)d_ws;
  size_t o = 0;
  auto alloc = [&](size_t n) -> size_t {
    o = (o + 255) & ~(size_t)255;
    size_t r = o; o += n; return r;
  };

  float*     xc32     = (float*)(ws + alloc((size_t)RC * Ll * 4));
  _Float16*  xc16     = (_Float16*)(ws + alloc((size_t)RC * Ll * 2));
  float*     meanb    = (float*)(ws + alloc(RC * 4));
  float*     stdevb   = (float*)(ws + alloc(RC * 4));
  _Float16*  tw16     = (_Float16*)(ws + alloc((size_t)RT * Dd * 2));
  float*     mbin     = (float*)(ws + alloc((size_t)RT * 2 * DIc * 4));   // reused below
  float*     xmconv32 = (float*)(ws + alloc((size_t)RT * DIc * 4));       // reused below
  _Float16*  xmconv16 = (_Float16*)(ws + alloc((size_t)RT * DIc * 2));
  float*     wxb      = (float*)(ws + alloc((size_t)RT * WXW * 4));
  float*     dtb      = (float*)(ws + alloc((size_t)RT * DIc * 4));       // reused below
  _Float16*  yg16     = (_Float16*)(ws + alloc((size_t)RT * DIc * 2));

  // Aliases into regions whose producers/consumers have finished by then:
  float*     twe0_32  = mbin;                                                  // 16.8 MB
  _Float16*  twe0_16  = (_Float16*)((char*)mbin + (size_t)RT * Dd * 4);        //  8.4 MB
  float*     ffnhid32 = (float*)((char*)mbin + (size_t)RT * Dd * 4 + (size_t)RT * Dd * 2);
  _Float16*  ffnhid16 = (_Float16*)xmconv32;                                   // 16.8 of 33.6 MB
  float*     twe32    = dtb;                                                   // 16.8 of 33.6 MB

  float*     cw32     = (float*)(ws + alloc((size_t)RC * Dd * 4));
  _Float16*  cw16     = (_Float16*)(ws + alloc((size_t)RC * Dd * 2));
  float*     zxb      = (float*)(ws + alloc((size_t)RC * ZXW * 4));
  float*     xbcb     = (float*)(ws + alloc((size_t)RC * XBCW * 4));
  float*     yhb      = (float*)(ws + alloc((size_t)RC * DIc * 4));
  _Float16*  yhn16    = (_Float16*)(ws + alloc((size_t)RC * DIc * 2));
  float*     cwe0_32  = (float*)(ws + alloc((size_t)RC * Dd * 4));
  _Float16*  cwe0_16  = (_Float16*)(ws + alloc((size_t)RC * Dd * 2));
  float*     cfhid32  = (float*)(ws + alloc((size_t)RC * DFFc * 4));
  _Float16*  cfhid16  = (_Float16*)(ws + alloc((size_t)RC * DFFc * 2));
  float*     cwe32    = (float*)(ws + alloc((size_t)RC * Dd * 4));
  _Float16*  cwe16    = (_Float16*)(ws + alloc((size_t)RC * Dd * 2));
  float*     filmb    = (float*)(ws + alloc((size_t)RC * 2 * Dd * 4));
  _Float16*  fused16  = (_Float16*)(ws + alloc((size_t)RC * NFc * 2));
  float*     headout  = (float*)(ws + alloc((size_t)RC * PREDc * 4));

  auto cvt = [&](const float* src, size_t n) -> _Float16* {
    _Float16* dst = (_Float16*)(ws + alloc(n * 2));
    k_cvt<<<dim3((unsigned)((n + 255) / 256)), dim3(256), 0, stream>>>(src, dst, (int)n);
    return dst;
  };
  _Float16* wWin   = cvt(mb_Win,  (size_t)2 * DIc * Dd);
  _Float16* wWx    = cvt(mb_Wx,   (size_t)WXW * DIc);
  _Float16* wWout  = cvt(mb_Wout, (size_t)Dd * DIc);
  _Float16* wtfW1  = cvt(tf_W1,   (size_t)DFFc * Dd);
  _Float16* wtfW2  = cvt(tf_W2,   (size_t)Dd * DFFc);
  _Float16* wWchan = cvt(W_chan,  (size_t)Dd * Ll);
  _Float16* wHyWin = cvt(hy_Win,  (size_t)ZXW * Dd);
  _Float16* wHyWo  = cvt(hy_Wout, (size_t)Dd * DIc);
  _Float16* wcfW1  = cvt(cf_W1,   (size_t)DFFc * Dd);
  _Float16* wcfW2  = cvt(cf_W2,   (size_t)Dd * DFFc);
  _Float16* wFilm  = cvt(film_W,  (size_t)2 * Dd * Dd);
  _Float16* wHead  = cvt(head_W,  (size_t)PREDc * NFc);

  auto gemm = [&](const _Float16* A, const _Float16* Bw, const float* bias,
                  const float* res, float* C, _Float16* C16,
                  int M, int N, int K, int act) {
    dim3 grid((unsigned)(M / 64), (unsigned)((N + 63) / 64));
    k_wmma_gemm<<<grid, dim3(128), 0, stream>>>(A, Bw, bias, res, C, C16, M, N, K, act);
  };

  // ---- token path -------------------------------------------------------
  k_norm<<<dim3(RC), dim3(256), 0, stream>>>(x_enc, xc32, xc16, meanb, stdevb);
  k_patch<<<dim3(RT), dim3(Dd), 0, stream>>>(xc32, W_patch, b_patch, tw16);
  gemm(tw16, wWin, nullptr, nullptr, mbin, nullptr, RT, 2 * DIc, Dd, 0);       // mb_Win
  k_dconv_m<<<dim3(RT), dim3(DIc), 0, stream>>>(mbin, mb_conv, mb_convb,
                                                xmconv32, xmconv16);
  gemm(xmconv16, wWx, nullptr, nullptr, wxb, nullptr, RT, WXW, DIc, 0);        // mb_Wx
  k_dtproj<<<dim3(RT), dim3(DIc), 0, stream>>>(wxb, mb_Wdt, mb_bdt, dtb);
  k_scan_m<<<dim3(RC), dim3(DIc), 0, stream>>>(xmconv32, dtb, wxb, mbin,
                                               mb_Alog, mb_D, yg16);
  gemm(yg16, wWout, nullptr, nullptr, twe0_32, twe0_16, RT, Dd, DIc, 0);       // mb_Wout
  gemm(twe0_16, wtfW1, tf_b1, nullptr, ffnhid32, ffnhid16, RT, DFFc, Dd, 1);   // FFN W1+gelu
  gemm(ffnhid16, wtfW2, tf_b2, twe0_32, twe32, nullptr, RT, Dd, DFFc, 0);      // FFN W2+res

  // ---- channel path -----------------------------------------------------
  gemm(xc16, wWchan, b_chan, nullptr, cw32, cw16, RC, Dd, Ll, 0);              // W_chan
  gemm(cw16, wHyWin, nullptr, nullptr, zxb, nullptr, RC, ZXW, Dd, 0);          // hy_Win
  k_dconv_h<<<dim3(RC), dim3(XBCW), 0, stream>>>(zxb, hy_conv, hy_convb, xbcb);
  k_hy_scan<<<dim3(Bb), dim3(Hh * HDc), 0, stream>>>(zxb, xbcb, hy_Alog,
                                                     hy_bdt, hy_D, yhb);
  k_rms<<<dim3(RC), dim3(DIc), 0, stream>>>(yhb, hy_normw, yhn16);
  gemm(yhn16, wHyWo, nullptr, nullptr, cwe0_32, cwe0_16, RC, Dd, DIc, 0);      // hy_Wout
  gemm(cwe0_16, wcfW1, cf_b1, nullptr, cfhid32, cfhid16, RC, DFFc, Dd, 1);     // cf W1+gelu
  gemm(cfhid16, wcfW2, cf_b2, cwe0_32, cwe32, cwe16, RC, Dd, DFFc, 0);         // cf W2+res
  gemm(cwe16, wFilm, film_b, nullptr, filmb, nullptr, RC, 2 * Dd, Dd, 0);      // FiLM

  // ---- fuse + head ------------------------------------------------------
  k_fuse<<<dim3((RC * NFc) / 256), dim3(256), 0, stream>>>(filmb, twe32, fused16);
  gemm(fused16, wHead, head_b, nullptr, headout, nullptr, RC, PREDc, NFc, 0);  // head
  k_final<<<dim3((RC * PREDc + 255) / 256), dim3(256), 0, stream>>>(headout, meanb,
                                                                    stdevb, out);
}
